// CrossAttention_3341484556491
// MI455X (gfx1250) — matrix operand order repaired and checked by reading, not compiled, not run on hardware
//
#include <hip/hip_runtime.h>
#include <cstdint>
#include <cstddef>

// ---------------------------------------------------------------------------
// CDNA5 (gfx1250) cross-attention, bf16 WMMA pipeline.
//   q = x@Wq, k = cond@Wk, v = cond@Wv         (bf16 WMMA GEMMs, f32 accum)
//   attn = softmax(q k^T * 0.125)              (flash-style, WMMA both einsums)
//   out  = (attn @ v) @ Wo + bo                (bf16 WMMA GEMM, fp32 out)
// ---------------------------------------------------------------------------

typedef __bf16 bf16_t;
typedef __attribute__((ext_vector_type(16))) __bf16 v16bf;
typedef __attribute__((ext_vector_type(8)))  float  v8f;

#define BATCH   4
#define NQ      2048
#define NK      2048
#define DMODEL  1024
#define HEADS   8
#define DHEAD   128
#define SCALE_F 0.125f            // DIM_HEAD=64 -> 64^-0.5
#define M_TOT   (BATCH * NQ)      // 8192

// ---- WMMA fragment helpers (ISA 7.12.2 layouts, wave32) --------------------
// A 16x32 bf16: lane<16 holds row=lane, K {0..7,16..23}; lane>=16 row=lane-16,
// K {8..15,24..31}. Two 16B chunks per lane.
__device__ __forceinline__ v16bf frag_a_load(const bf16_t* base, int ldk, int lane) {
  const int row = lane & 15;
  const int k1  = (lane < 16) ? 0 : 8;
  const bf16_t* p = base + row * ldk + k1;
  union { v16bf v; uint4 u[2]; } f;
  f.u[0] = *(const uint4*)(p);
  f.u[1] = *(const uint4*)(p + 16);
  return f.v;
}

// B 32x16 bf16 sourced from Bt[n][k] (row-major along K): lane<16 holds
// column n=lane, K {0..7,16..23}; lane>=16 column n=lane-16, K {8..15,24..31}.
__device__ __forceinline__ v16bf frag_b_load(const bf16_t* base, int ldk, int lane) {
  const int n  = lane & 15;
  const int k1 = (lane < 16) ? 0 : 8;
  const bf16_t* p = base + n * ldk + k1;
  union { v16bf v; uint4 u[2]; } f;
  f.u[0] = *(const uint4*)(p);
  f.u[1] = *(const uint4*)(p + 16);
  return f.v;
}

__device__ __forceinline__ v8f wmma_bf16(v16bf a, v16bf b, v8f c) {
  return __builtin_amdgcn_wmma_f32_16x16x32_bf16(false, a, false, b,
                                                 (short)0, c, false, false);
}

// ---- prep: fp32 -> bf16 (4-wide) ------------------------------------------
__global__ void cvt_f32_bf16(const float* __restrict__ src,
                             bf16_t* __restrict__ dst, int n4) {
  int i = blockIdx.x * blockDim.x + threadIdx.x;
  if (i >= n4) return;
  float4 f = ((const float4*)src)[i];
  union { bf16_t h[4]; uint2 u; } o;
  o.h[0] = (bf16_t)f.x; o.h[1] = (bf16_t)f.y;
  o.h[2] = (bf16_t)f.z; o.h[3] = (bf16_t)f.w;
  ((uint2*)dst)[i] = o.u;
}

// ---- prep: weight transpose + convert: Wt[n][k] = bf16(W[k][n]) -----------
__global__ void wtrans_f32_bf16(const float* __restrict__ W,
                                bf16_t* __restrict__ Wt) {
  int idx = blockIdx.x * blockDim.x + threadIdx.x;  // idx = n*1024 + k
  int n = idx >> 10;
  int k = idx & 1023;
  Wt[idx] = (bf16_t)W[k * DMODEL + n];
}

// ---- GEMM: C[8192x1024] = A[8192x1024(bf16)] x Bt[1024x1024(bf16, NxK)] ---
// MODE 0: bf16 row-major store (Q,K projections)
// MODE 1: bf16 transposed-per-(b,h) store for V: Vt[b][h][d][j]
// MODE 2: fp32 store + bias (final output projection -> d_out)
template <int MODE>
__global__ __launch_bounds__(256, 1)
void gemm_bf16(const bf16_t* __restrict__ A, const bf16_t* __restrict__ Bt,
               bf16_t* __restrict__ Cb, float* __restrict__ Cf,
               const float* __restrict__ bias) {
  constexpr int BK = 32, BM = 128, BN = 128;
  constexpr int Kdim = 1024, Ndim = 1024;
  __shared__ bf16_t sA[BM * BK];   // 8 KB
  __shared__ bf16_t sB[BN * BK];   // 8 KB
  const int tid  = threadIdx.x;
  const int lane = tid & 31;
  const int w    = tid >> 5;       // 8 waves
  const int wm   = w & 3;          // wave row 0..3  (32 rows each)
  const int wn   = w >> 2;         // wave col 0..1  (64 cols each)
  const int m0   = blockIdx.y * BM;
  const int n0   = blockIdx.x * BN;

  const v8f vzero = {0.f, 0.f, 0.f, 0.f, 0.f, 0.f, 0.f, 0.f};
  v8f acc[2][4];
#pragma unroll
  for (int mt = 0; mt < 2; ++mt)
#pragma unroll
    for (int nt = 0; nt < 4; ++nt) acc[mt][nt] = vzero;

  for (int kt = 0; kt < Kdim; kt += BK) {
    __syncthreads();
    // cooperative fill: 512 16B chunks per tile, 2 per thread per tile
#pragma unroll
    for (int c = 0; c < 2; ++c) {
      int chunk = tid * 2 + c;          // 0..511
      int row   = chunk >> 2;           // 0..127
      int koff  = (chunk & 3) * 8;      // elems within 32-wide K slab
      *(uint4*)(sA + row * BK + koff) =
          *(const uint4*)(A + (size_t)(m0 + row) * Kdim + kt + koff);
      *(uint4*)(sB + row * BK + koff) =
          *(const uint4*)(Bt + (size_t)(n0 + row) * Kdim + kt + koff);
      if (kt + BK < Kdim) {             // prefetch next K slab
        __builtin_prefetch(A + (size_t)(m0 + row) * Kdim + kt + BK + koff, 0, 1);
        __builtin_prefetch(Bt + (size_t)(n0 + row) * Kdim + kt + BK + koff, 0, 1);
      }
    }
    __syncthreads();

    v16bf af[2];
#pragma unroll
    for (int mt = 0; mt < 2; ++mt)
      af[mt] = frag_a_load(sA + (wm * 32 + mt * 16) * BK, BK, lane);
#pragma unroll
    for (int nt = 0; nt < 4; ++nt) {
      v16bf bfr = frag_b_load(sB + (wn * 64 + nt * 16) * BK, BK, lane);
#pragma unroll
      for (int mt = 0; mt < 2; ++mt)
        acc[mt][nt] = wmma_bf16(af[mt], bfr, acc[mt][nt]);
    }
  }

  // epilogue: C 16x16 layout -> lane<16: row=r, col=lane; lane>=16: row=8+r
  const int rowoff = (lane < 16) ? 0 : 8;
  const int col    = lane & 15;
#pragma unroll
  for (int mt = 0; mt < 2; ++mt)
#pragma unroll
    for (int nt = 0; nt < 4; ++nt)
#pragma unroll
      for (int r = 0; r < 8; ++r) {
        float val = acc[mt][nt][r];
        int gm = m0 + wm * 32 + mt * 16 + r + rowoff;
        int gn = n0 + wn * 64 + nt * 16 + col;
        if constexpr (MODE == 0) {
          Cb[(size_t)gm * Ndim + gn] = (bf16_t)val;
        } else if constexpr (MODE == 1) {
          int b = gm >> 11, j = gm & 2047;
          int h = gn >> 7,  dd = gn & 127;
          Cb[(((size_t)((b * HEADS + h) * DHEAD + dd)) << 11) + j] = (bf16_t)val;
        } else {
          Cf[(size_t)gm * Ndim + gn] = val + bias[gn];
        }
      }
}

// ---- flash attention: per wave 16 q-rows, online softmax over Nk ----------
__global__ __launch_bounds__(256, 1)
void attn_kernel(const bf16_t* __restrict__ Q, const bf16_t* __restrict__ K,
                 const bf16_t* __restrict__ Vt, bf16_t* __restrict__ O) {
  __shared__ bf16_t sP[8][16 * 32];   // per-wave private P staging (8 KB)
  const int lane = threadIdx.x & 31;
  const int w    = threadIdx.x >> 5;
  const int b    = blockIdx.z;
  const int h    = blockIdx.y;
  const int q0   = blockIdx.x * 128 + w * 16;

  // Q fragments for this wave's 16 rows, whole 128-dim head, kept in VGPRs.
  const bf16_t* qbase = Q + (size_t)(b * NQ + q0) * DMODEL + h * DHEAD;
  v16bf qf[4];
#pragma unroll
  for (int kk = 0; kk < 4; ++kk)
    qf[kk] = frag_a_load(qbase + kk * 32, DMODEL, lane);

  const v8f vzero = {0.f, 0.f, 0.f, 0.f, 0.f, 0.f, 0.f, 0.f};
  float mrow[8], lrow[8];
  v8f o[8];
#pragma unroll
  for (int r = 0; r < 8; ++r) { mrow[r] = -1e30f; lrow[r] = 0.f; }
#pragma unroll
  for (int t = 0; t < 8; ++t) o[t] = vzero;

  const bf16_t* kbase = K + (size_t)(b * NK) * DMODEL + h * DHEAD;
  const bf16_t* vbase = Vt + (size_t)((b * HEADS + h) * DHEAD) * NK;
  const int rowoff = (lane < 16) ? 0 : 8;
  const int col    = lane & 15;

  for (int j0 = 0; j0 < NK; j0 += 32) {
    // S = Q K^T over 32 keys: 2 n-tiles x 4 k-steps
    v8f s[2];
    s[0] = vzero; s[1] = vzero;
#pragma unroll
    for (int kk = 0; kk < 4; ++kk)
#pragma unroll
      for (int nt = 0; nt < 2; ++nt) {
        v16bf kf = frag_b_load(
            kbase + (size_t)(j0 + nt * 16) * DMODEL + kk * 32, DMODEL, lane);
        s[nt] = wmma_bf16(qf[kk], kf, s[nt]);
      }

    // online softmax: row r of a lane-half lives in VGPR r across 16 lanes
    float p0[8], p1[8];
#pragma unroll
    for (int r = 0; r < 8; ++r) {
      float s0 = s[0][r] * SCALE_F;
      float s1 = s[1][r] * SCALE_F;
      float cur = fmaxf(s0, s1);
#pragma unroll
      for (int off = 1; off < 16; off <<= 1)
        cur = fmaxf(cur, __shfl_xor(cur, off, 32));
      float mnew  = fmaxf(mrow[r], cur);
      float alpha = __expf(mrow[r] - mnew);
      p0[r] = __expf(s0 - mnew);
      p1[r] = __expf(s1 - mnew);
      float rs = p0[r] + p1[r];
#pragma unroll
      for (int off = 1; off < 16; off <<= 1)
        rs += __shfl_xor(rs, off, 32);
      lrow[r] = lrow[r] * alpha + rs;
      mrow[r] = mnew;
#pragma unroll
      for (int t = 0; t < 8; ++t) o[t][r] *= alpha;
    }

    // reshape P (C-layout) -> A-fragment layout via wave-private LDS
#pragma unroll
    for (int r = 0; r < 8; ++r) {
      sP[w][(r + rowoff) * 32 + col]      = (bf16_t)p0[r];
      sP[w][(r + rowoff) * 32 + 16 + col] = (bf16_t)p1[r];
    }
    asm volatile("s_wait_dscnt 0" ::: "memory");  // LDS RAW, same wave
    v16bf pf = frag_a_load(&sP[w][0], 32, lane);

    // O += P @ V, V kept transposed (Vt[d][j]) so B-frags are contiguous
#pragma unroll
    for (int t = 0; t < 8; ++t) {
      v16bf vf = frag_b_load(vbase + (size_t)(t * 16) * NK + j0, NK, lane);
      o[t] = wmma_bf16(pf, vf, o[t]);
    }
  }

  // normalize and store bf16 (row-major [8192][1024], col = h*128 + d)
  float inv[8];
#pragma unroll
  for (int r = 0; r < 8; ++r) inv[r] = 1.0f / lrow[r];
#pragma unroll
  for (int t = 0; t < 8; ++t)
#pragma unroll
    for (int r = 0; r < 8; ++r) {
      int gm = b * NQ + q0 + r + rowoff;
      int gn = h * DHEAD + t * 16 + col;
      O[(size_t)gm * DMODEL + gn] = (bf16_t)(o[t][r] * inv[r]);
    }
}

// ---------------------------------------------------------------------------
extern "C" void kernel_launch(void* const* d_in, const int* in_sizes, int n_in,
                              void* d_out, int out_size, void* d_ws,
                              size_t ws_size, hipStream_t stream) {
  (void)in_sizes; (void)n_in; (void)out_size; (void)ws_size;
  const float* x    = (const float*)d_in[0];
  const float* cond = (const float*)d_in[1];
  const float* Wq   = (const float*)d_in[2];
  const float* Wk   = (const float*)d_in[3];
  const float* Wv   = (const float*)d_in[4];
  const float* Wo   = (const float*)d_in[5];
  const float* bo   = (const float*)d_in[6];
  float* out = (float*)d_out;

  // workspace carve-up (~104 MB total)
  char* ws = (char*)d_ws;
  size_t off = 0;
  auto carve = [&](size_t bytes) -> char* {
    char* p = ws + off;
    off += (bytes + 255) & ~(size_t)255;
    return p;
  };
  const size_t act = (size_t)M_TOT * DMODEL * sizeof(bf16_t);  // 16 MB
  const size_t wgt = (size_t)DMODEL * DMODEL * sizeof(bf16_t); //  2 MB
  bf16_t* xb  = (bf16_t*)carve(act);
  bf16_t* cb  = (bf16_t*)carve(act);
  bf16_t* wqt = (bf16_t*)carve(wgt);
  bf16_t* wkt = (bf16_t*)carve(wgt);
  bf16_t* wvt = (bf16_t*)carve(wgt);
  bf16_t* wot = (bf16_t*)carve(wgt);
  bf16_t* qb  = (bf16_t*)carve(act);
  bf16_t* kb  = (bf16_t*)carve(act);
  bf16_t* vt  = (bf16_t*)carve(act);   // [b][h][d][j]
  bf16_t* ob  = (bf16_t*)carve(act);

  // 1) convert activations to bf16
  const int n4 = M_TOT * DMODEL / 4;   // 2M float4s
  cvt_f32_bf16<<<n4 / 256, 256, 0, stream>>>(x, xb, n4);
  cvt_f32_bf16<<<n4 / 256, 256, 0, stream>>>(cond, cb, n4);

  // 2) transpose+convert weights to Bt[n][k] bf16
  const int nw = DMODEL * DMODEL;      // 1M
  wtrans_f32_bf16<<<nw / 256, 256, 0, stream>>>(Wq, wqt);
  wtrans_f32_bf16<<<nw / 256, 256, 0, stream>>>(Wk, wkt);
  wtrans_f32_bf16<<<nw / 256, 256, 0, stream>>>(Wv, wvt);
  wtrans_f32_bf16<<<nw / 256, 256, 0, stream>>>(Wo, wot);

  // 3) projections (WMMA GEMMs)
  dim3 gg(DMODEL / 128, M_TOT / 128);  // (8, 64)
  gemm_bf16<0><<<gg, 256, 0, stream>>>(xb, wqt, qb, nullptr, nullptr);
  gemm_bf16<0><<<gg, 256, 0, stream>>>(cb, wkt, kb, nullptr, nullptr);
  gemm_bf16<1><<<gg, 256, 0, stream>>>(cb, wvt, vt, nullptr, nullptr);

  // 4) attention (flash-style, WMMA)
  attn_kernel<<<dim3(NQ / 128, HEADS, BATCH), 256, 0, stream>>>(qb, kb, vt, ob);

  // 5) output projection + bias -> fp32 d_out
  gemm_bf16<2><<<gg, 256, 0, stream>>>(ob, wot, nullptr, out, bo);
}
